// AttentionOT_33191507263875
// MI455X (gfx1250) — matrix operand order repaired and checked by reading, not compiled, not run on hardware
//
#include <hip/hip_runtime.h>

// ---------- types ----------
typedef _Float16 half16 __attribute__((ext_vector_type(16)));
typedef _Float16 half8  __attribute__((ext_vector_type(8)));
typedef float    f32x8  __attribute__((ext_vector_type(8)));
typedef int      i32x4v __attribute__((vector_size(16)));

// shapes (fixed by reference)
#define NQ_  64
#define M_   16
#define B_   8
#define K_   256
#define C_   512

#define EPS_OT    0.05f
#define INV_EPS   20.0f
#define LOG_MU   (-2.77258856f)   // log(1/16 + 1e-8)
#define LOG_NU   (-4.15888245f)   // log(1/64 + 1e-8)

#if defined(__has_builtin)
#if __has_builtin(__builtin_amdgcn_global_load_async_to_lds_b128)
#define HAS_ASYNC_LDS 1
#else
#define HAS_ASYNC_LDS 0
#endif
#else
#define HAS_ASYNC_LDS 0
#endif

#if HAS_ASYNC_LDS
__device__ __forceinline__ void async_copy16(const _Float16* g, _Float16* l) {
  __builtin_amdgcn_global_load_async_to_lds_b128(
      (__attribute__((address_space(1))) i32x4v*)(g),
      (__attribute__((address_space(3))) i32x4v*)(l), 0, 0);
}
template <int N>
__device__ __forceinline__ void wait_async() {
#if __has_builtin(__builtin_amdgcn_s_wait_asynccnt)
  __builtin_amdgcn_s_wait_asynccnt(N);
#else
  asm volatile("s_wait_asynccnt %0" ::"i"(N) : "memory");
#endif
}
#endif

// ---------- elementwise convert f32 -> f16 ----------
__global__ __launch_bounds__(256) void cvt_f16(const float* __restrict__ in,
                                               _Float16* __restrict__ out, int n) {
  int id = blockIdx.x * 256 + threadIdx.x;
  if (id < n) out[id] = (_Float16)in[id];
}

// ---------- remap xq (Nq,M,B,C) -> per-batch (B, 1024, C) with r = m*64+n, f16 ----------
__global__ __launch_bounds__(256) void remap_xq(const float* __restrict__ in,
                                                _Float16* __restrict__ out, int n) {
  int id = blockIdx.x * 256 + threadIdx.x;
  if (id >= n) return;
  int b   = id >> 19;           // 1024*512 = 2^19
  int rem = id & ((1 << 19) - 1);
  int r   = rem >> 9;           // C = 512 = 2^9
  int c   = rem & 511;
  int nq  = r & 63;
  int mq  = r >> 6;
  long long src = ((long long)(nq * M_ + mq) * B_ + b) * C_ + c;
  out[id] = (_Float16)in[src];
}

// ---------- generic WMMA GEMM: C[M,N] = A[M,Kd] * B[N,Kd]^T  (f16 in, f32 acc) ----------
// 128 threads (4 waves) per block; block tile 64x64, K-step 32.
// Async global->LDS double-buffered staging of A/B tiles (CDNA5 path).
// MODE 0: write f32 (ldc=N, batch stride sC)
// MODE 1: write f16 (ldc=N, batch stride sC)
// MODE 2: write f32 to d_out with (n,m,b) row scatter + bias
template <int MODE>
__global__ __launch_bounds__(128) void gemm_wmma_kernel(
    const _Float16* __restrict__ A, long long sA,
    const _Float16* __restrict__ Bm, long long sB,
    void* __restrict__ Cout, long long sC,
    int Kd, int N, const float* __restrict__ bias) {
  const int tid  = threadIdx.x;
  const int wv   = tid >> 5;
  const int lane = tid & 31;
  const int rlo  = lane & 15;   // A row within tile / B column within tile / D column
  const int hi   = lane >> 4;   // half-wave selector
  const int rowBase = blockIdx.y * 64;
  const int colBase = blockIdx.x * 64;
  const int bz = blockIdx.z;

  const _Float16* Ab = A  + (long long)bz * sA;
  const _Float16* Bb = Bm + (long long)bz * sB;

  f32x8 acc[4] = {};

#if HAS_ASYNC_LDS
  __shared__ __align__(16) _Float16 As[2][64 * 32];
  __shared__ __align__(16) _Float16 Bs[2][64 * 32];
  const int cr  = tid >> 2;        // 0..31: row within 32-row chunk
  const int seg = (tid & 3) * 8;   // 16B segment within a 64B row

  auto issue = [&](int buf, int k0) {
#pragma unroll
    for (int inst = 0; inst < 2; ++inst) {
      const int r = inst * 32 + cr;
      async_copy16(Ab + (long long)(rowBase + r) * Kd + k0 + seg, &As[buf][r * 32 + seg]);
      async_copy16(Bb + (long long)(colBase + r) * Kd + k0 + seg, &Bs[buf][r * 32 + seg]);
    }
  };

  const int nsteps = Kd >> 5;
  issue(0, 0);
  for (int s = 0; s < nsteps; ++s) {
    const int buf = s & 1;
    __syncthreads();                       // B1: prefetch target buffer free
    if (s + 1 < nsteps) {
      issue(buf ^ 1, (s + 1) * 32);
      wait_async<4>();                     // own current-buffer ops complete
    } else {
      wait_async<0>();
    }
    __syncthreads();                       // B2: everyone's current-buffer ops complete

    // Load A frag + all four B frags first, then issue WMMAs back-to-back.
    union { half16 v; half8 h[2]; } af, bf[4];
    const int ar = wv * 16 + rlo;
    af.h[0] = *(const half8*)&As[buf][ar * 32 + hi * 16];
    af.h[1] = *(const half8*)&As[buf][ar * 32 + hi * 16 + 8];
#pragma unroll
    for (int t = 0; t < 4; ++t) {
      const int br = t * 16 + rlo;
      bf[t].h[0] = *(const half8*)&Bs[buf][br * 32 + hi * 16];
      bf[t].h[1] = *(const half8*)&Bs[buf][br * 32 + hi * 16 + 8];
    }
#pragma unroll
    for (int t = 0; t < 4; ++t)
      acc[t] = __builtin_amdgcn_wmma_f32_16x16x32_f16(
          false, af.v, false, bf[t].v, (short)0, acc[t], false, false);
  }
#else
  // fallback: direct global loads per wave
  const _Float16* Arow = Ab + (long long)(rowBase + wv * 16 + rlo) * Kd;
  const int aOff = hi * 16;
  const int bOff = hi * 16;
  for (int k0 = 0; k0 < Kd; k0 += 32) {
    union { half16 v; half8 h[2]; } af, bf[4];
    af.h[0] = *(const half8*)(Arow + k0 + aOff);
    af.h[1] = *(const half8*)(Arow + k0 + aOff + 8);
#pragma unroll
    for (int t = 0; t < 4; ++t) {
      const _Float16* Brow = Bb + (long long)(colBase + t * 16 + rlo) * Kd + k0 + bOff;
      bf[t].h[0] = *(const half8*)(Brow);
      bf[t].h[1] = *(const half8*)(Brow + 8);
    }
#pragma unroll
    for (int t = 0; t < 4; ++t)
      acc[t] = __builtin_amdgcn_wmma_f32_16x16x32_f16(
          false, af.v, false, bf[t].v, (short)0, acc[t], false, false);
  }
#endif

  // D layout: lane holds column (rlo), rows hi*8 + i
  const int r0 = rowBase + wv * 16 + hi * 8;
  const int c0 = colBase + rlo;
  if (MODE == 0) {
    float* C = (float*)Cout + (long long)bz * sC;
#pragma unroll
    for (int t = 0; t < 4; ++t)
#pragma unroll
      for (int i = 0; i < 8; ++i)
        C[(long long)(r0 + i) * N + (c0 + t * 16)] = acc[t][i];
  } else if (MODE == 1) {
    _Float16* C = (_Float16*)Cout + (long long)bz * sC;
#pragma unroll
    for (int t = 0; t < 4; ++t)
#pragma unroll
      for (int i = 0; i < 8; ++i)
        C[(long long)(r0 + i) * N + (c0 + t * 16)] = (_Float16)acc[t][i];
  } else {
    float* C = (float*)Cout;
#pragma unroll
    for (int t = 0; t < 4; ++t) {
      const int c = c0 + t * 16;
      const float bv = bias[c];
#pragma unroll
      for (int i = 0; i < 8; ++i) {
        int r  = r0 + i;
        int nq = r & 63, mq = r >> 6;
        long long grow = (long long)(nq * M_ + mq) * B_ + bz;  // (n,m,b)
        C[grow * C_ + c] = acc[t][i] + bv;
      }
    }
  }
}

// ---------- row l2norm (+f16), optional transpose for V: out (B, C, K) ----------
__global__ __launch_bounds__(256) void rownorm_kernel(const float* __restrict__ X,
                                                      _Float16* __restrict__ Y,
                                                      int rows, int doNorm, int transposeV) {
  const int wid  = threadIdx.x >> 5;
  const int lane = threadIdx.x & 31;
  const int row  = blockIdx.x * 8 + wid;
  if (row >= rows) return;
  const float* xr = X + (long long)row * C_;
  float vals[16];
  float ss = 0.0f;
#pragma unroll
  for (int i = 0; i < 16; ++i) {
    vals[i] = xr[lane + i * 32];
    ss += vals[i] * vals[i];
  }
#pragma unroll
  for (int mask = 16; mask >= 1; mask >>= 1) ss += __shfl_xor(ss, mask);
  const float rn = doNorm ? (1.0f / fmaxf(sqrtf(ss), 1e-12f)) : 1.0f;
  if (!transposeV) {
    _Float16* yr = Y + (long long)row * C_;
#pragma unroll
    for (int i = 0; i < 16; ++i) yr[lane + i * 32] = (_Float16)(vals[i] * rn);
  } else {
    const int b = row >> 8;        // rows-per-batch = 256
    const int k = row & 255;
#pragma unroll
    for (int i = 0; i < 16; ++i) {
      int c = lane + i * 32;
      Y[((long long)b * C_ + c) * K_ + k] = (_Float16)(vals[i] * rn);
    }
  }
}

// ---------- Sinkhorn: one wave per (b,k) 16x64 problem, fully in registers ----------
__global__ __launch_bounds__(256) void sinkhorn_kernel(const float* __restrict__ Sim,
                                                       float* __restrict__ score,
                                                       _Float16* __restrict__ Tt) {
  const int p    = blockIdx.x * 8 + (threadIdx.x >> 5);  // (b*256 + k), 2048 total
  const int lane = threadIdx.x & 31;
  const int mr   = lane & 15;  // row m
  const int h    = lane >> 4;  // column half: cols [h*32, h*32+32)
  const long long off = (long long)p * 1024 + mr * 64 + h * 32;

  // A = -wdist/eps = (sim - 1)/eps
  float A[32];
  const float4* src = (const float4*)(Sim + off);
#pragma unroll
  for (int q = 0; q < 8; ++q) {
    float4 t = src[q];
    A[q * 4 + 0] = (t.x - 1.0f) * INV_EPS;
    A[q * 4 + 1] = (t.y - 1.0f) * INV_EPS;
    A[q * 4 + 2] = (t.z - 1.0f) * INV_EPS;
    A[q * 4 + 3] = (t.w - 1.0f) * INV_EPS;
  }
  float Vv[32];
#pragma unroll
  for (int j = 0; j < 32; ++j) Vv[j] = 0.0f;
  float U = 0.0f;

#pragma unroll 1
  for (int it = 0; it < 100; ++it) {
    // u-update: U = log_mu - LSE_j(A + V)   (old U cancels analytically)
    float m = A[0] + Vv[0];
#pragma unroll
    for (int j = 1; j < 32; ++j) m = fmaxf(m, A[j] + Vv[j]);
    m = fmaxf(m, __shfl_xor(m, 16));
    float s = 0.0f;
#pragma unroll
    for (int j = 0; j < 32; ++j) s += __expf(A[j] + Vv[j] - m);
    s += __shfl_xor(s, 16);
    U = LOG_MU - m - __logf(s);

    // v-update: V_j = log_nu - LSE_i(A + U_new), reduce over 16 rows (xor 1,2,4,8)
    float cm[32];
#pragma unroll
    for (int j = 0; j < 32; ++j) cm[j] = A[j] + U;
#pragma unroll
    for (int mask = 1; mask <= 8; mask <<= 1)
#pragma unroll
      for (int j = 0; j < 32; ++j) cm[j] = fmaxf(cm[j], __shfl_xor(cm[j], mask));
    float cs[32];
#pragma unroll
    for (int j = 0; j < 32; ++j) cs[j] = __expf(A[j] + U - cm[j]);
#pragma unroll
    for (int mask = 1; mask <= 8; mask <<= 1)
#pragma unroll
      for (int j = 0; j < 32; ++j) cs[j] += __shfl_xor(cs[j], mask);
#pragma unroll
    for (int j = 0; j < 32; ++j) Vv[j] = LOG_NU - cm[j] - __logf(cs[j]);
  }

  // emit score_map and T^T (f16, (B, 1024, K)) for the attn GEMM
  const int bq = p >> 8, kq = p & 255;
  float* sc = score + off;
#pragma unroll
  for (int q = 0; q < 8; ++q) {
    float sv[4];
#pragma unroll
    for (int e = 0; e < 4; ++e) {
      int j = q * 4 + e;
      float T = __expf(A[j] + U + Vv[j]);
      sv[e] = 1024.0f * (1.0f + EPS_OT * A[j]) * T;  // M*Nq*sim*T, sim = 1 + eps*A
      long long r = mr * 64 + h * 32 + j;
      Tt[((long long)bq * 1024 + r) * K_ + kq] = (_Float16)T;
    }
    ((float4*)sc)[q] = make_float4(sv[0], sv[1], sv[2], sv[3]);
  }
}

// ---------- launch ----------
extern "C" void kernel_launch(void* const* d_in, const int* in_sizes, int n_in,
                              void* d_out, int out_size, void* d_ws, size_t ws_size,
                              hipStream_t stream) {
  const float* xq = (const float*)d_in[0];
  const float* xk = (const float*)d_in[1];
  const float* xv = (const float*)d_in[2];
  const float* Wq = (const float*)d_in[3];
  const float* Wk = (const float*)d_in[4];
  const float* Wv = (const float*)d_in[5];
  const float* Wp = (const float*)d_in[6];
  const float* bp = (const float*)d_in[7];

  char* w = (char*)d_ws;
  // f16 weights: 4 x 512*512*2B
  _Float16* Wqh = (_Float16*)(w + 0);
  _Float16* Wkh = (_Float16*)(w + 524288);
  _Float16* Wvh = (_Float16*)(w + 1048576);
  _Float16* Wph = (_Float16*)(w + 1572864);
  // f16 inputs
  _Float16* XQh = (_Float16*)(w + 2097152);    // (B,1024,512)  8 MB
  _Float16* XKh = (_Float16*)(w + 10485760);   // (B,256,512)   2 MB
  _Float16* XVh = (_Float16*)(w + 12582912);   // (B,256,512)   2 MB
  // f16 normalized / transposed
  _Float16* Qn  = (_Float16*)(w + 14680064);   // (B,1024,512)  8 MB
  _Float16* Kn  = (_Float16*)(w + 23068672);   // (B,256,512)   2 MB
  _Float16* VfT = (_Float16*)(w + 25165824);   // (B,512,256)   2 MB
  // big region: raw f32 GEMM outputs, later aliased by Sim/Tt/Xpre
  char* big = w + 27262976;                    // 24 MB region
  float*    Qraw = (float*)(big);              // 16 MB
  float*    Kraw = (float*)(big + 16777216);   //  4 MB
  float*    Vraw = (float*)(big + 20971520);   //  4 MB
  float*    Sim  = (float*)(big);              //  8 MB (aliases dead Qraw)
  _Float16* Tt   = (_Float16*)(big + 8388608); //  4 MB (B,1024,256)
  _Float16* Xpre = (_Float16*)(big + 12582912);//  8 MB (B,1024,512)

  float* out_x     = (float*)d_out;            // (Nq,M,B,C) = 4194304 f32
  float* out_score = out_x + 4194304;          // (B,K,M,Nq) = 2097152 f32

  // 1) converts / remap
  cvt_f16<<<1024, 256, 0, stream>>>(Wq, Wqh, 262144);
  cvt_f16<<<1024, 256, 0, stream>>>(Wk, Wkh, 262144);
  cvt_f16<<<1024, 256, 0, stream>>>(Wv, Wvh, 262144);
  cvt_f16<<<1024, 256, 0, stream>>>(Wp, Wph, 262144);
  cvt_f16<<<4096, 256, 0, stream>>>(xk, XKh, 1048576);
  cvt_f16<<<4096, 256, 0, stream>>>(xv, XVh, 1048576);
  remap_xq<<<16384, 256, 0, stream>>>(xq, XQh, 4194304);

  // 2) projections (f16 WMMA, f32 out)
  gemm_wmma_kernel<0><<<dim3(8, 16, 8), 128, 0, stream>>>(
      XQh, 1024LL * 512, Wqh, 0LL, (void*)Qraw, 1024LL * 512, 512, 512, nullptr);
  gemm_wmma_kernel<0><<<dim3(8, 4, 8), 128, 0, stream>>>(
      XKh, 256LL * 512, Wkh, 0LL, (void*)Kraw, 256LL * 512, 512, 512, nullptr);
  gemm_wmma_kernel<0><<<dim3(8, 4, 8), 128, 0, stream>>>(
      XVh, 256LL * 512, Wvh, 0LL, (void*)Vraw, 256LL * 512, 512, 512, nullptr);

  // 3) l2norm Q,K -> f16 ; V -> f16 transposed (B,C,K)
  rownorm_kernel<<<1024, 256, 0, stream>>>(Qraw, Qn, 8192, 1, 0);
  rownorm_kernel<<<256, 256, 0, stream>>>(Kraw, Kn, 2048, 1, 0);
  rownorm_kernel<<<256, 256, 0, stream>>>(Vraw, VfT, 2048, 0, 1);

  // 4) sim[b] (256 x 1024) = Kn[b] @ Qn[b]^T
  gemm_wmma_kernel<0><<<dim3(16, 4, 8), 128, 0, stream>>>(
      Kn, 256LL * 512, Qn, 1024LL * 512, (void*)Sim, 256LL * 1024, 512, 1024, nullptr);

  // 5) Sinkhorn (100 iters), writes score_map + T^T
  sinkhorn_kernel<<<256, 256, 0, stream>>>(Sim, out_score, Tt);

  // 6) x_pre[b] (1024 x 512) = Tt[b] @ VfT[b]^T  (contraction over k=256)
  gemm_wmma_kernel<1><<<dim3(8, 16, 8), 128, 0, stream>>>(
      Tt, 1024LL * 256, VfT, 512LL * 256, (void*)Xpre, 1024LL * 512, 256, 512, nullptr);

  // 7) out = x_pre @ Wp^T + bp, scattered to (Nq,M,B,C)
  gemm_wmma_kernel<2><<<dim3(8, 16, 8), 128, 0, stream>>>(
      Xpre, 1024LL * 512, Wph, 0LL, (void*)out_x, 0LL, 512, 512, bp);
}
